// NetworkRNN_reimplement_28192165331633
// MI455X (gfx1250) — matrix operand order repaired and checked by reading, not compiled, not run on hardware
//
#include <hip/hip_runtime.h>
#include <hip/hip_bf16.h>

// Problem constants from the reference
#define B_   64
#define S_   512
#define E_   256
#define H_   512
#define NWG  16          // workgroups cooperating on the recurrence
#define LDS_STRIDE 520   // 512 + 8 bf16 pad -> row stride = 1040B = 4-bank shift

typedef __attribute__((ext_vector_type(16))) __bf16 v16bf;
typedef __attribute__((ext_vector_type(8)))  __bf16 bf16x8;
typedef __attribute__((ext_vector_type(8)))  float  v8f;
typedef __attribute__((ext_vector_type(4)))  float  f32x4;

// Build a 16-bit WMMA operand fragment (16 bf16 = 8 VGPRs) from two
// 8-element contiguous chunks. Elements 0..7 <- p0[0..7], 8..15 <- p1[0..7].
static __device__ inline v16bf load_frag_f32(const float* p0, const float* p1) {
  f32x4 x0 = *(const f32x4*)(p0);
  f32x4 x1 = *(const f32x4*)(p0 + 4);
  f32x4 y0 = *(const f32x4*)(p1);
  f32x4 y1 = *(const f32x4*)(p1 + 4);
  v16bf a;
#pragma unroll
  for (int i = 0; i < 4; ++i) {
    a[i]      = (__bf16)x0[i];
    a[i + 4]  = (__bf16)x1[i];
    a[i + 8]  = (__bf16)y0[i];
    a[i + 12] = (__bf16)y1[i];
  }
  return a;
}

static __device__ inline v16bf load_frag_bf16(const __bf16* p0, const __bf16* p1) {
  bf16x8 lo = *(const bf16x8*)p0;   // 16B contiguous
  bf16x8 hi = *(const bf16x8*)p1;   // 16B contiguous
  v16bf a;
#pragma unroll
  for (int i = 0; i < 8; ++i) { a[i] = lo[i]; a[i + 8] = hi[i]; }
  return a;
}

// ---------------------------------------------------------------------------
// Kernel 1: fused embedding gather + input projection.
//   X[(s*B+b)*H + h] = emb_table[words[b,s]] . Wi[h,:] + bi[h] + bh[h]
// (both biases folded in; fp32 result feeds the recurrence accumulator)
// One wave per 16x16 output tile; K=256 -> 8 bf16 WMMAs.
// ---------------------------------------------------------------------------
__global__ __launch_bounds__(256)
void k_embed_xproj(const int* __restrict__ words, const float* __restrict__ emb,
                   const float* __restrict__ Wi, const float* __restrict__ bi,
                   const float* __restrict__ bh, float* __restrict__ X) {
  const int lane = threadIdx.x & 31;
  const int wave = threadIdx.x >> 5;
  const int tile = blockIdx.x * 8 + wave;        // 2048 * 32 tiles
  const int nt = tile & 31;                      // H/16 = 32 column tiles
  const int mt = tile >> 5;                      // (S*B)/16 = 2048 row tiles
  const int mbase = mt * 16, nbase = nt * 16;

  // A operand: gathered embedding row (row index m = s*B + b)
  const int arow_m = mbase + (lane & 15);
  const int s = arow_m >> 6;                     // B_ = 64
  const int b = arow_m & 63;
  const int word = words[b * S_ + s];
  const float* arow = emb + (size_t)word * E_;
  const int koff = (lane >> 4) * 8;              // A: upper lanes hold K+8 chunk

  // B operand: Wi^T -> lane = output column, contiguous K in memory
  const int ncol = nbase + (lane & 15);
  const float* brow = Wi + (size_t)ncol * E_;
  const int kb = (lane >> 4) * 16;               // unused: B loads with koff, same k order as A

  v8f acc = {};
#pragma unroll
  for (int kk = 0; kk < E_ / 32; ++kk) {
    const int k0 = kk * 32;
    v16bf a  = load_frag_f32(arow + k0 + koff, arow + k0 + 16 + koff);
    v16bf bm = load_frag_f32(brow + k0 + koff, brow + k0 + 16 + koff);
    acc = __builtin_amdgcn_wmma_f32_16x16x32_bf16(false, a, false, bm,
                                                  (short)0, acc, false, false);
  }

  const float bias = bi[ncol] + bh[ncol];
#pragma unroll
  for (int r = 0; r < 8; ++r) {                  // C/D layout: M = r + 8*(lane>>4)
    const int m = mbase + r + 8 * (lane >> 4);
    X[(size_t)m * H_ + ncol] = acc[r] + bias;
  }
}

// ---------------------------------------------------------------------------
// Init kernel: zero h0 and the inter-step barrier counter (graph-replay safe).
// ---------------------------------------------------------------------------
__global__ void k_init(__bf16* __restrict__ h0buf, unsigned* __restrict__ bar) {
  const int idx = blockIdx.x * blockDim.x + threadIdx.x;
  if (idx < B_ * H_) h0buf[idx] = (__bf16)0.0f;
  if (idx == 0) *bar = 0u;
}

// ---------------------------------------------------------------------------
// Kernel 2: persistent recurrence over 512 serial steps + final projection.
// 16 workgroups; WG g owns output columns [g*32, g*32+32). Its Wh slice lives
// in LDS (bf16) for the whole kernel. Each step, the full 64KB hidden state
// is bulk-staged into LDS with GLOBAL_LOAD_ASYNC_TO_LDS_B128 (ASYNCcnt path),
// so the K-loop runs entirely out of LDS. Hidden state ping-pongs in L2;
// steps separated by an agent-scope release/acquire counter barrier.
// ---------------------------------------------------------------------------
__global__ __launch_bounds__(256)
void k_rnn(const float* __restrict__ Wh, const float* __restrict__ X,
           const float* __restrict__ Wf, const float* __restrict__ bfin,
           __bf16* __restrict__ h0buf, __bf16* __restrict__ h1buf,
           unsigned* __restrict__ bar, float* __restrict__ out) {
  __shared__ __bf16 ldsWh[32 * LDS_STRIDE];      // 33,280 B: resident Wh slice
  __shared__ __bf16 ldsH [B_ * LDS_STRIDE];      // 66,560 B: staged h_t

  const int g = blockIdx.x;
  const int colbase = g * 32;

  // Preload this WGP's 32 rows of Wh (= its output columns) as bf16 into LDS.
  for (int idx = threadIdx.x; idx < 32 * H_; idx += 256) {
    const int lj = idx >> 9;                     // H_ = 512
    const int i  = idx & (H_ - 1);
    ldsWh[lj * LDS_STRIDE + i] = (__bf16)Wh[(size_t)(colbase + lj) * H_ + i];
  }
  __syncthreads();

  const int lane = threadIdx.x & 31;
  const int wave = threadIdx.x >> 5;
  const int rt = wave >> 1;                      // 4 row tiles (B = 64)
  const int ct = wave & 1;                       // 2 local column tiles
  const int mbase  = rt * 16;
  const int nlocal = ct * 16 + (lane & 15);
  const int jglob  = colbase + nlocal;
  const int koff = (lane >> 4) * 8;
  const int kb   = (lane >> 4) * 16;
  const int arow = mbase + (lane & 15);
  const int mofs = 8 * (lane >> 4);

  for (int t = 0; t < S_; ++t) {
    const __bf16* hsrc = (t & 1) ? h1buf : h0buf;
    __bf16*       hdst = (t & 1) ? h0buf : h1buf;

    // --- Async bulk-stage h_t (64x512 bf16 = 64KB) from L2 into padded LDS.
    // 256 threads x 16 iters x 16B = 65,536B; tracked with ASYNCcnt.
#pragma unroll
    for (int it = 0; it < 16; ++it) {
      const int elem = (it * 256 + threadIdx.x) * 8;       // 8 bf16 per lane
      const int row = elem >> 9;                           // /512
      const int col = elem & (H_ - 1);
      const unsigned lds_addr = (unsigned)(size_t)&ldsH[row * LDS_STRIDE + col];
      const __bf16* gp = hsrc + elem;
      asm volatile("global_load_async_to_lds_b128 %0, %1, off"
                   :: "v"(lds_addr), "v"(gp) : "memory");
    }

    // Accumulator initialized with x_t + bi + bh (pre-folded fp32 in X);
    // overlaps with the async staging above.
    v8f acc;
    const float* xcol = X + (size_t)t * B_ * H_ + jglob;
#pragma unroll
    for (int r = 0; r < 8; ++r)
      acc[r] = xcol[(size_t)(mbase + r + mofs) * H_];

    asm volatile("s_wait_asynccnt 0x0" ::: "memory");
    __syncthreads();                              // staged h visible to all waves

    const __bf16* ap = ldsH  + (size_t)arow   * LDS_STRIDE;  // A: staged h_t
    const __bf16* bp = ldsWh + (size_t)nlocal * LDS_STRIDE;  // B: resident Wh
#pragma unroll
    for (int kk = 0; kk < H_ / 32; ++kk) {
      const int k0 = kk * 32;
      v16bf a  = load_frag_bf16(ap + k0 + koff, ap + k0 + 16 + koff);
      v16bf bm = load_frag_bf16(bp + k0 + koff, bp + k0 + 16 + koff);
      acc = __builtin_amdgcn_wmma_f32_16x16x32_bf16(false, a, false, bm,
                                                    (short)0, acc, false, false);
    }

    // Pull next step's x_t stream toward the caches (HBM-streamed, 128KB/step)
    if (t + 1 < S_)
      __builtin_prefetch(X + (size_t)(t + 1) * B_ * H_ + (size_t)mbase * H_ + jglob, 0, 1);

#pragma unroll
    for (int r = 0; r < 8; ++r) {
      const int m = mbase + r + mofs;
      const float sg = 1.0f / (1.0f + __expf(-acc[r]));
      hdst[(size_t)m * H_ + jglob] = (__bf16)sg;
      if (t == S_ - 1) out[B_ + (size_t)m * H_ + jglob] = sg;  // fp32 hidden
    }

    // Inter-workgroup step barrier: monotonic counter, release/acquire at
    // agent scope so the bf16 hidden slices are visible across WGPs.
    __syncthreads();
    if (threadIdx.x == 0) {
      __hip_atomic_fetch_add(bar, 1u, __ATOMIC_RELEASE, __HIP_MEMORY_SCOPE_AGENT);
      const unsigned target = (unsigned)(t + 1) * NWG;
      while (__hip_atomic_load(bar, __ATOMIC_ACQUIRE, __HIP_MEMORY_SCOPE_AGENT) < target)
        __builtin_amdgcn_s_sleep(1);
    }
    __syncthreads();
  }

  // Final head: sig[b] = sigmoid(hidden[b,:] . Wf + bf), O = 1 (tiny)
  if (g == 0 && threadIdx.x < B_) {
    const int b = threadIdx.x;
    float a = bfin[0];
    const float* hr = out + B_ + (size_t)b * H_;
    for (int i = 0; i < H_; ++i) a += hr[i] * Wf[i];
    out[b] = 1.0f / (1.0f + __expf(-a));
  }
}

// ---------------------------------------------------------------------------
extern "C" void kernel_launch(void* const* d_in, const int* in_sizes, int n_in,
                              void* d_out, int out_size, void* d_ws, size_t ws_size,
                              hipStream_t stream) {
  (void)in_sizes; (void)n_in; (void)out_size; (void)ws_size;
  const int*   words = (const int*)  d_in[0];   // input_words [B,S]
  const float* emb   = (const float*)d_in[1];   // emb_table  [V,E]
  const float* Wh    = (const float*)d_in[2];   // [H,H]
  const float* bh    = (const float*)d_in[3];   // [H]
  const float* Wi    = (const float*)d_in[4];   // [H,E]
  const float* bi    = (const float*)d_in[5];   // [H]
  const float* Wf    = (const float*)d_in[6];   // [O,H] = [1,512]
  const float* bfin  = (const float*)d_in[7];   // [1]
  float* out = (float*)d_out;                   // [64 sig | 64*512 hidden]

  // Workspace layout
  char* ws = (char*)d_ws;
  float* X = (float*)ws;                                      // 64 MB fp32 xproj
  size_t off = (size_t)S_ * B_ * H_ * sizeof(float);
  __bf16* h0buf = (__bf16*)(ws + off); off += (size_t)B_ * H_ * sizeof(__bf16);
  __bf16* h1buf = (__bf16*)(ws + off); off += (size_t)B_ * H_ * sizeof(__bf16);
  unsigned* bar = (unsigned*)(ws + off);

  k_init<<<(B_ * H_ + 255) / 256, 256, 0, stream>>>(h0buf, bar);
  k_embed_xproj<<<(2048 * 32) / 8, 256, 0, stream>>>(words, emb, Wi, bi, bh, X);
  k_rnn<<<NWG, 256, 0, stream>>>(Wh, X, Wf, bfin, h0buf, h1buf, bar, out);
}
